// ParallelSARDecoder_PT_35167192220319
// MI455X (gfx1250) — matrix operand order repaired and checked by reading, not compiled, not run on hardware
//
#include <hip/hip_runtime.h>
#include <hip/hip_bf16.h>
#include <math.h>

// ---------------------------------------------------------------------------
// ParallelSARDecoder forward for MI455X (gfx1250, wave32, WMMA).
// All dense algebra goes through v_wmma_f32_16x16x32_bf16 (f32 accumulate).
// Sequential LSTM recurrence = 41 stream-ordered kernel launches per layer.
// ---------------------------------------------------------------------------

typedef __bf16 bf16;
typedef __attribute__((ext_vector_type(16))) __bf16 v16bf;
typedef __attribute__((ext_vector_type(8)))  float  v8f;

#define S_LEN 41
#define N_B   64
#define DH    512
#define G4    2048
#define DK    64
#define DM    512
#define HH    8
#define WW    40
#define HW    320
#define T_LEN 40
#define NCM1  92
#define KCONV 4608   // DM*9
#define COMBW 1536   // DH + DM + DH

// ---------------- WMMA fragment helpers (gfx1250 layouts) -------------------

// A: 16x32 bf16 (MxK). lanes 0-15 row m, K in {0..7, 16..23}; lanes 16-31 same
// row, K in {8..15, 24..31}. Row clamped for M-guard (results masked at store).
__device__ __forceinline__ v16bf load_A(const bf16* A, long lda, int row0,
                                        int k0, int Mvalid) {
  int lane = threadIdx.x & 31;
  int m = row0 + (lane & 15);
  if (m >= Mvalid) m = Mvalid - 1;
  int khalf = (lane >> 4) * 8;
  const bf16* p = A + (long)m * lda + k0 + khalf;
  v16bf a;
#pragma unroll
  for (int j = 0; j < 8; ++j)  a[j] = p[j];
#pragma unroll
  for (int j = 8; j < 16; ++j) a[j] = p[j + 8];
  return a;
}

// B: 32x16 bf16 (KxN), sourced from row-major weight Wm[n][k] (B = Wm^T).
// lane n holds column n; lanes 0-15 K in {0..7, 16..23}, lanes 16-31
// K in {8..15, 24..31} (same order as A).
__device__ __forceinline__ v16bf load_B(const bf16* Wm, long ldw, int n0,
                                        int k0, int Nvalid) {
  int lane = threadIdx.x & 31;
  int n = n0 + (lane & 15);
  if (n >= Nvalid) n = Nvalid - 1;
  int kh = (lane >> 4) * 8;
  const bf16* p = Wm + (long)n * ldw + k0 + kh;
  v16bf b;
#pragma unroll
  for (int j = 0; j < 16; ++j) b[j] = p[j + ((j >= 8) ? 8 : 0)];
  return b;
}

__device__ __forceinline__ v8f wmma_bf16(v16bf a, v16bf b, v8f c) {
  return __builtin_amdgcn_wmma_f32_16x16x32_bf16(
      /*neg_a=*/false, a, /*neg_b=*/false, b,
      /*c_mod=*/(short)0, c, /*reuse_a=*/false, /*reuse_b=*/false);
}

__device__ __forceinline__ float sigmoidf_(float x) {
  return 1.0f / (1.0f + __expf(-x));
}

// ---------------- elementwise utility kernels -------------------------------

__global__ void k_f2bf(const float* __restrict__ x, bf16* __restrict__ y, long n) {
  long i = (long)blockIdx.x * blockDim.x + threadIdx.x;
  if (i < n) y[i] = (bf16)x[i];
}

__global__ void k_zero_f(float* x, long n) {
  long i = (long)blockIdx.x * blockDim.x + threadIdx.x;
  if (i < n) x[i] = 0.0f;
}

__global__ void k_zero_bf(bf16* x, long n) {
  long i = (long)blockIdx.x * blockDim.x + threadIdx.x;
  if (i < n) x[i] = (bf16)0.0f;
}

// dec_in[b, s, d] : s==0 -> holistic[b, d] ; s>0 -> emb[labels[b, s-1], d]
__global__ void k_build_decin(const float* __restrict__ hol,
                              const int* __restrict__ labels,
                              const float* __restrict__ emb,
                              bf16* __restrict__ decin) {
  long i = (long)blockIdx.x * blockDim.x + threadIdx.x;
  long tot = (long)N_B * S_LEN * DH;
  if (i >= tot) return;
  int d = (int)(i % DH);
  long r = i / DH;
  int s = (int)(r % S_LEN);
  int b = (int)(r / S_LEN);
  float v;
  if (s == 0) v = hol[(long)b * DH + d];
  else {
    int lab = labels[(long)b * T_LEN + (s - 1)];
    v = emb[(long)lab * DH + d];
  }
  decin[i] = (bf16)v;
}

// ---------------- generic GEMM: C[M,Nn] = A[M,K] * W[Nn,K]^T (+bias) --------

__global__ void k_gemm(const bf16* __restrict__ A, long lda,
                       const bf16* __restrict__ Wm, long ldw,
                       const float* __restrict__ bias0,
                       const float* __restrict__ bias1,
                       float* __restrict__ C, long ldc,
                       int M, int Nn, int K) {
  int n0 = blockIdx.x * 16;
  int r0 = blockIdx.y * 16;
  v8f acc = {};
  for (int k = 0; k < K; k += 32) {
    v16bf a = load_A(A, lda, r0, k, M);
    v16bf b = load_B(Wm, ldw, n0, k, Nn);
    acc = wmma_bf16(a, b, acc);
  }
  int lane = threadIdx.x & 31;
  int nn = lane & 15, mb = (lane >> 4) * 8;
  int ncol = n0 + nn;
  float badd = 0.0f;
  if (ncol < Nn) {
    if (bias0) badd += bias0[ncol];
    if (bias1) badd += bias1[ncol];
  }
#pragma unroll
  for (int v = 0; v < 8; ++v) {
    int row = r0 + mb + v;
    if (row < M && ncol < Nn)
      C[(long)row * ldc + ncol] = acc[v] + badd;
  }
}

// ---------------- fused LSTM step -------------------------------------------
// One block per (16 batch rows x 16 hidden cols). Computes the 4 gate tiles of
// hprev @ Whh^T (K=512, A-fragment shared by all 4 gates), adds precomputed xg,
// applies gate math in-register on the WMMA accumulator layout, updates c, and
// writes h in bf16 straight into the output sequence (next step's A operand).

__global__ void k_lstm_step(const bf16* __restrict__ hprev, long ldh,
                            const float* __restrict__ xgt, long ldxg,
                            const bf16* __restrict__ Whh,
                            float* __restrict__ cst,
                            bf16* __restrict__ hout, long ldo) {
  int jc = blockIdx.x * 16;  // column within DH
  int rb = blockIdx.y * 16;  // batch rows
  v8f acc0 = {}, acc1 = {}, acc2 = {}, acc3 = {};
  for (int k = 0; k < DH; k += 32) {
    v16bf a = load_A(hprev, ldh, rb, k, N_B);
    v16bf b0 = load_B(Whh, DH, 0 * DH + jc, k, G4);
    acc0 = wmma_bf16(a, b0, acc0);
    v16bf b1 = load_B(Whh, DH, 1 * DH + jc, k, G4);
    acc1 = wmma_bf16(a, b1, acc1);
    v16bf b2 = load_B(Whh, DH, 2 * DH + jc, k, G4);
    acc2 = wmma_bf16(a, b2, acc2);
    v16bf b3 = load_B(Whh, DH, 3 * DH + jc, k, G4);
    acc3 = wmma_bf16(a, b3, acc3);
  }
  int lane = threadIdx.x & 31;
  int nn = lane & 15, mb = (lane >> 4) * 8;
  int col = jc + nn;
#pragma unroll
  for (int v = 0; v < 8; ++v) {
    int b = rb + mb + v;  // batch index
    const float* xr = xgt + (long)b * ldxg;
    float gi = acc0[v] + xr[0 * DH + col];
    float gf = acc1[v] + xr[1 * DH + col];
    float gg = acc2[v] + xr[2 * DH + col];
    float go = acc3[v] + xr[3 * DH + col];
    float cc = sigmoidf_(gf) * cst[(long)b * DH + col] +
               sigmoidf_(gi) * tanhf(gg);
    cst[(long)b * DH + col] = cc;
    float hh = sigmoidf_(go) * tanhf(cc);
    hout[(long)b * ldo + col] = (bf16)hh;
  }
}

// ---------------- 3x3 SAME conv as im2col GEMM ------------------------------
// rows = b*HW + h*WW + w (M = 20480), K = DM*9 ordered (c, ky, kx) to match
// row-major Wk[DK, DM, 3, 3].

__device__ __forceinline__ v16bf load_A_conv(const bf16* __restrict__ feat,
                                             int row0, int k0) {
  int lane = threadIdx.x & 31;
  int row = row0 + (lane & 15);
  int b = row / HW;
  int hw = row % HW;
  int h = hw / WW, w = hw % WW;
  int khalf = (lane >> 4) * 8;
  v16bf a;
#pragma unroll
  for (int j = 0; j < 16; ++j) {
    int kk = k0 + khalf + j + ((j >= 8) ? 8 : 0);
    int c = kk / 9, r = kk % 9;
    int hy = h + r / 3 - 1;
    int wx = w + r % 3 - 1;
    bf16 val = (bf16)0.0f;
    if (hy >= 0 && hy < HH && wx >= 0 && wx < WW)
      val = feat[((long)b * DM + c) * HW + hy * WW + wx];
    a[j] = val;
  }
  return a;
}

__global__ void k_conv_gemm(const bf16* __restrict__ feat,
                            const bf16* __restrict__ Wk,
                            const float* __restrict__ bk,
                            float* __restrict__ kout) {
  int n0 = blockIdx.x * 16;  // DK
  int r0 = blockIdx.y * 16;  // N_B*HW
  v8f acc = {};
  for (int k = 0; k < KCONV; k += 32) {
    v16bf a = load_A_conv(feat, r0, k);
    v16bf b = load_B(Wk, KCONV, n0, k, DK);
    acc = wmma_bf16(a, b, acc);
  }
  int lane = threadIdx.x & 31;
  int nn = lane & 15, mb = (lane >> 4) * 8;
  int ncol = n0 + nn;
  float badd = bk[ncol];
#pragma unroll
  for (int v = 0; v < 8; ++v) {
    int row = r0 + mb + v;
    kout[(long)row * DK + ncol] = acc[v] + badd;
  }
}

// ---------------- attention energy + valid-width mask -----------------------

__global__ void k_energy(const float* __restrict__ kout,
                         const float* __restrict__ q,
                         const float* __restrict__ We,
                         const float* __restrict__ be,
                         const float* __restrict__ vr,
                         float* __restrict__ energy) {
  long i = (long)blockIdx.x * blockDim.x + threadIdx.x;
  long tot = (long)N_B * S_LEN * HW;
  if (i >= tot) return;
  int hw = (int)(i % HW);
  long r = i / HW;
  int s = (int)(r % S_LEN);
  int b = (int)(r / S_LEN);
  int w = hw % WW;
  float vw = ceilf(vr[b] * (float)WW);
  if (vw > (float)WW) vw = (float)WW;
  if ((float)w >= vw) { energy[i] = -INFINITY; return; }
  const float* kr = kout + ((long)b * HW + hw) * DK;
  const float* qr = q + ((long)b * S_LEN + s) * DK;
  float e = be[0];
#pragma unroll 4
  for (int kk = 0; kk < DK; ++kk) e += We[kk] * tanhf(kr[kk] + qr[kk]);
  energy[i] = e;
}

// ---------------- row softmax over HW (one block per (b,s)) -----------------

__global__ void k_softmax(const float* __restrict__ energy,
                          bf16* __restrict__ attn) {
  int row = blockIdx.x;  // b*S_LEN + s
  const float* e = energy + (long)row * HW;
  __shared__ float red[128];
  int t = threadIdx.x;
  float m = -INFINITY;
  for (int i = t; i < HW; i += 128) m = fmaxf(m, e[i]);
  red[t] = m; __syncthreads();
  for (int o = 64; o > 0; o >>= 1) {
    if (t < o) red[t] = fmaxf(red[t], red[t + o]);
    __syncthreads();
  }
  m = red[0]; __syncthreads();
  float ssum = 0.0f;
  for (int i = t; i < HW; i += 128) ssum += expf(e[i] - m);
  red[t] = ssum; __syncthreads();
  for (int o = 64; o > 0; o >>= 1) {
    if (t < o) red[t] += red[t + o];
    __syncthreads();
  }
  float inv = 1.0f / red[0];
  for (int i = t; i < HW; i += 128)
    attn[(long)row * HW + i] = (bf16)(expf(e[i] - m) * inv);
}

// ---------------- batched attn-feat GEMM -> comb[:,512:1024] ----------------
// attn_feat[b,s,c] = sum_hw attn[b,s,hw] * feat[b,c,hw]; K=HW=320.

__global__ void k_attnfeat(const bf16* __restrict__ attn,
                           const bf16* __restrict__ feat,
                           bf16* __restrict__ comb) {
  int b = blockIdx.z;
  int n0 = blockIdx.x * 16;  // DM
  int r0 = blockIdx.y * 16;  // S (padded to 48)
  const bf16* A = attn + (long)b * S_LEN * HW;
  const bf16* Wm = feat + (long)b * DM * HW;
  v8f acc = {};
  for (int k = 0; k < HW; k += 32) {
    v16bf a = load_A(A, HW, r0, k, S_LEN);
    v16bf bb = load_B(Wm, HW, n0, k, DM);
    acc = wmma_bf16(a, bb, acc);
  }
  int lane = threadIdx.x & 31;
  int nn = lane & 15, mb = (lane >> 4) * 8;
  int ncol = n0 + nn;
#pragma unroll
  for (int v = 0; v < 8; ++v) {
    int s = r0 + mb + v;
    if (s < S_LEN)
      comb[((long)b * S_LEN + s) * COMBW + DH + ncol] = (bf16)acc[v];
  }
}

// comb[:,0:512] = y1 ; comb[:,1024:1536] = holistic (broadcast over s)
__global__ void k_pack_comb(const bf16* __restrict__ y1,
                            const bf16* __restrict__ hol_bf,
                            bf16* __restrict__ comb) {
  long i = (long)blockIdx.x * blockDim.x + threadIdx.x;
  long tot = (long)N_B * S_LEN * DH;
  if (i >= tot) return;
  int d = (int)(i % DH);
  long r = i / DH;
  int b = (int)(r / S_LEN);
  comb[r * COMBW + d] = y1[i];
  comb[r * COMBW + 2 * DH + d] = hol_bf[(long)b * DH + d];
}

// ---------------- logits GEMM, store [:,1:,:] slice -------------------------

__global__ void k_logits(const bf16* __restrict__ comb,
                         const bf16* __restrict__ Wp,
                         const float* __restrict__ bp,
                         float* __restrict__ out) {
  int n0 = blockIdx.x * 16;  // NCM1 padded to 96
  int r0 = blockIdx.y * 16;  // N_B*S_LEN = 2624
  v8f acc = {};
  for (int k = 0; k < COMBW; k += 32) {
    v16bf a = load_A(comb, COMBW, r0, k, N_B * S_LEN);
    v16bf b = load_B(Wp, COMBW, n0, k, NCM1);
    acc = wmma_bf16(a, b, acc);
  }
  int lane = threadIdx.x & 31;
  int nn = lane & 15, mb = (lane >> 4) * 8;
  int ncol = n0 + nn;
#pragma unroll
  for (int v = 0; v < 8; ++v) {
    int row = r0 + mb + v;
    int b = row / S_LEN, s = row % S_LEN;
    if (ncol < NCM1 && s >= 1)
      out[((long)b * T_LEN + (s - 1)) * NCM1 + ncol] = acc[v] + bp[ncol];
  }
}

// ---------------------------------------------------------------------------

extern "C" void kernel_launch(void* const* d_in, const int* in_sizes, int n_in,
                              void* d_out, int out_size, void* d_ws, size_t ws_size,
                              hipStream_t stream) {
  (void)in_sizes; (void)n_in; (void)out_size; (void)ws_size;

  const float* feat     = (const float*)d_in[0];   // [N,DM,H,W]
  const float* holistic = (const float*)d_in[1];   // [N,DH]
  const int*   labels   = (const int*)  d_in[2];   // [N,T]
  const float* vr       = (const float*)d_in[3];   // [N]
  const float* emb      = (const float*)d_in[4];   // [NC,DH]
  const float* Wih0     = (const float*)d_in[5];
  const float* Whh0     = (const float*)d_in[6];
  const float* bih0     = (const float*)d_in[7];
  const float* bhh0     = (const float*)d_in[8];
  const float* Wih1     = (const float*)d_in[9];
  const float* Whh1     = (const float*)d_in[10];
  const float* bih1     = (const float*)d_in[11];
  const float* bhh1     = (const float*)d_in[12];
  const float* Wq       = (const float*)d_in[13];
  const float* bq       = (const float*)d_in[14];
  const float* Wk       = (const float*)d_in[15];
  const float* bk       = (const float*)d_in[16];
  const float* We       = (const float*)d_in[17];
  const float* be       = (const float*)d_in[18];
  const float* Wp       = (const float*)d_in[19];
  const float* bp       = (const float*)d_in[20];
  float* out = (float*)d_out;

  // ---- workspace carve-up (256B aligned) ----
  char* w = (char*)d_ws;
  auto alloc = [&](size_t bytes) -> void* {
    void* p = (void*)w;
    w += (bytes + 255) & ~(size_t)255;
    return p;
  };
  bf16* decin_bf = (bf16*)alloc((size_t)N_B * S_LEN * DH * 2);
  bf16* wih0_bf  = (bf16*)alloc((size_t)G4 * DH * 2);
  bf16* whh0_bf  = (bf16*)alloc((size_t)G4 * DH * 2);
  bf16* wih1_bf  = (bf16*)alloc((size_t)G4 * DH * 2);
  bf16* whh1_bf  = (bf16*)alloc((size_t)G4 * DH * 2);
  bf16* wq_bf    = (bf16*)alloc((size_t)DK * DH * 2);
  bf16* wk_bf    = (bf16*)alloc((size_t)DK * KCONV * 2);
  bf16* wp_bf    = (bf16*)alloc((size_t)NCM1 * COMBW * 2);
  bf16* feat_bf  = (bf16*)alloc((size_t)N_B * DM * HW * 2);
  bf16* hol_bf   = (bf16*)alloc((size_t)N_B * DH * 2);
  bf16* hzero_bf = (bf16*)alloc((size_t)N_B * DH * 2);
  float* xg      = (float*)alloc((size_t)N_B * S_LEN * G4 * 4);
  float* cst     = (float*)alloc((size_t)N_B * DH * 4);
  bf16* y0_bf    = (bf16*)alloc((size_t)N_B * S_LEN * DH * 2);
  bf16* y1_bf    = (bf16*)alloc((size_t)N_B * S_LEN * DH * 2);
  float* qbuf    = (float*)alloc((size_t)N_B * S_LEN * DK * 4);
  float* kout    = (float*)alloc((size_t)N_B * HW * DK * 4);
  float* energy  = (float*)alloc((size_t)N_B * S_LEN * HW * 4);
  bf16* attn_bf  = (bf16*)alloc((size_t)N_B * S_LEN * HW * 2);
  bf16* comb     = (bf16*)alloc((size_t)N_B * S_LEN * COMBW * 2);

  const int BT = 256;
  auto gl = [&](long n) { return dim3((unsigned)((n + BT - 1) / BT)); };

  // ---- bf16 conversions ----
  k_f2bf<<<gl((long)G4 * DH), BT, 0, stream>>>(Wih0, wih0_bf, (long)G4 * DH);
  k_f2bf<<<gl((long)G4 * DH), BT, 0, stream>>>(Whh0, whh0_bf, (long)G4 * DH);
  k_f2bf<<<gl((long)G4 * DH), BT, 0, stream>>>(Wih1, wih1_bf, (long)G4 * DH);
  k_f2bf<<<gl((long)G4 * DH), BT, 0, stream>>>(Whh1, whh1_bf, (long)G4 * DH);
  k_f2bf<<<gl((long)DK * DH), BT, 0, stream>>>(Wq, wq_bf, (long)DK * DH);
  k_f2bf<<<gl((long)DK * KCONV), BT, 0, stream>>>(Wk, wk_bf, (long)DK * KCONV);
  k_f2bf<<<gl((long)NCM1 * COMBW), BT, 0, stream>>>(Wp, wp_bf, (long)NCM1 * COMBW);
  k_f2bf<<<gl((long)N_B * DM * HW), BT, 0, stream>>>(feat, feat_bf, (long)N_B * DM * HW);
  k_f2bf<<<gl((long)N_B * DH), BT, 0, stream>>>(holistic, hol_bf, (long)N_B * DH);
  k_zero_bf<<<gl((long)N_B * DH), BT, 0, stream>>>(hzero_bf, (long)N_B * DH);
  k_build_decin<<<gl((long)N_B * S_LEN * DH), BT, 0, stream>>>(holistic, labels, emb, decin_bf);

  const int Mrows = N_B * S_LEN;              // 2624 (divisible by 16)
  const long ldy  = (long)S_LEN * DH;         // row stride of [b, s, d] seq views
  const long ldxg = (long)S_LEN * G4;

  // ---- layer 0: xg = dec_in @ Wih0^T + (bih0 + bhh0) ----
  k_gemm<<<dim3(G4 / 16, Mrows / 16), 32, 0, stream>>>(
      decin_bf, DH, wih0_bf, DH, bih0, bhh0, xg, G4, Mrows, G4, DH);
  k_zero_f<<<gl((long)N_B * DH), BT, 0, stream>>>(cst, (long)N_B * DH);
  for (int t = 0; t < S_LEN; ++t) {
    const bf16* hp = (t == 0) ? hzero_bf : (y0_bf + (long)(t - 1) * DH);
    long ldh = (t == 0) ? (long)DH : ldy;
    k_lstm_step<<<dim3(DH / 16, N_B / 16), 32, 0, stream>>>(
        hp, ldh, xg + (long)t * G4, ldxg, whh0_bf, cst, y0_bf + (long)t * DH, ldy);
  }

  // ---- layer 1 ----
  k_gemm<<<dim3(G4 / 16, Mrows / 16), 32, 0, stream>>>(
      y0_bf, DH, wih1_bf, DH, bih1, bhh1, xg, G4, Mrows, G4, DH);
  k_zero_f<<<gl((long)N_B * DH), BT, 0, stream>>>(cst, (long)N_B * DH);
  for (int t = 0; t < S_LEN; ++t) {
    const bf16* hp = (t == 0) ? hzero_bf : (y1_bf + (long)(t - 1) * DH);
    long ldh = (t == 0) ? (long)DH : ldy;
    k_lstm_step<<<dim3(DH / 16, N_B / 16), 32, 0, stream>>>(
        hp, ldh, xg + (long)t * G4, ldxg, whh1_bf, cst, y1_bf + (long)t * DH, ldy);
  }

  // ---- attention query / key ----
  k_gemm<<<dim3(DK / 16, Mrows / 16), 32, 0, stream>>>(
      y1_bf, DH, wq_bf, DH, bq, (const float*)nullptr, qbuf, DK, Mrows, DK, DH);
  k_conv_gemm<<<dim3(DK / 16, (N_B * HW) / 16), 32, 0, stream>>>(
      feat_bf, wk_bf, bk, kout);

  // ---- energy + mask, softmax ----
  k_energy<<<gl((long)N_B * S_LEN * HW), BT, 0, stream>>>(kout, qbuf, We, be, vr, energy);
  k_softmax<<<dim3(N_B * S_LEN), 128, 0, stream>>>(energy, attn_bf);

  // ---- attn-weighted features + concat + classifier ----
  k_attnfeat<<<dim3(DM / 16, 3 /*ceil(41/16)*/, N_B), 32, 0, stream>>>(
      attn_bf, feat_bf, comb);
  k_pack_comb<<<gl((long)N_B * S_LEN * DH), BT, 0, stream>>>(y1_bf, hol_bf, comb);
  k_logits<<<dim3(96 / 16, Mrows / 16), 32, 0, stream>>>(comb, wp_bf, bp, out);
}
